// fuse_extract_21586505630002
// MI455X (gfx1250) — matrix operand order repaired and checked by reading, not compiled, not run on hardware
//
#include <hip/hip_runtime.h>
#include <hip/hip_bf16.h>

typedef __attribute__((ext_vector_type(16))) _Float16 v16h;
typedef __attribute__((ext_vector_type(8)))  _Float16 v8h;
typedef __attribute__((ext_vector_type(8)))  float    v8f;
typedef __attribute__((ext_vector_type(4)))  int      v4i;

#define NOBJ 4
#define CIMG 512
#define HW   1024
#define DPTS 1024
#define L0   2048
#define KNN  16
#define BN_SCALE 0.9999950000374997f
#define EPSGN 1e-5f

// ----- CDNA5 async global->LDS path (guarded; falls back to regs+ds_store) -----
#if defined(__HIP_DEVICE_COMPILE__) && __has_builtin(__builtin_amdgcn_global_load_async_to_lds_b128)
#define USE_ASYNC_LDS 1
#else
#define USE_ASYNC_LDS 0
#endif

#if USE_ASYNC_LDS
typedef __attribute__((address_space(1))) v4i gv4i;   // global int4
typedef __attribute__((address_space(3))) v4i lv4i;   // LDS int4
static __device__ __forceinline__ gv4i* as1p(const void* p) {
  return (gv4i*)(unsigned long long)(__SIZE_TYPE__)p;
}
static __device__ __forceinline__ lv4i* as3p(void* p) {
  return (lv4i*)(unsigned int)(__SIZE_TYPE__)p;   // generic LDS VA low 32b == LDS offset
}
static __device__ __forceinline__ void wait_async0() {
#if __has_builtin(__builtin_amdgcn_s_wait_asynccnt)
  __builtin_amdgcn_s_wait_asynccnt(0);
#else
  asm volatile("s_wait_asynccnt 0" ::: "memory");
#endif
}
#endif

// ---------------- tiny utility kernels ----------------

__global__ void cast_f16_k(const float* __restrict__ src, _Float16* __restrict__ dst, int n) {
  int i = blockIdx.x * 256 + threadIdx.x;
  if (i < n) dst[i] = (_Float16)src[i];
}

// imgT[(obj*1024+g)*512 + c] = imgs[(obj*512+c)*1024 + g]
__global__ void transpose_img_k(const float* __restrict__ imgs, _Float16* __restrict__ imgT) {
  int i = blockIdx.x * 256 + threadIdx.x;        // 4*1024*512
  int c = i & 511, g = (i >> 9) & 1023, obj = i >> 19;
  imgT[i] = (_Float16)imgs[((obj << 9) + c) * 1024 + g];
}

// ptsT[(obj*2048+l)*1024 + c] = points[c*8192 + obj*2048 + l]
__global__ void transpose_pts_k(const float* __restrict__ points, _Float16* __restrict__ ptsT) {
  int i = blockIdx.x * 256 + threadIdx.x;        // 4*2048*1024
  int c = i & 1023, l = (i >> 10) & 2047, obj = i >> 21;
  ptsT[i] = (_Float16)points[(size_t)c * 8192 + obj * 2048 + l];
}

__global__ void project_k(const float* __restrict__ xys, const float* __restrict__ M,
                          float* __restrict__ proj) {
  int i = blockIdx.x * 256 + threadIdx.x;        // 4*2048
  int obj = i >> 11, l = i & 2047;
  float x = xys[obj * 2048 + l];
  float y = xys[8192 + obj * 2048 + l];
  const float* m = M + obj * 9;
  proj[i * 2 + 0] = m[0] * x + m[1] * y + m[2];
  proj[i * 2 + 1] = m[3] * x + m[4] * y + m[5];
}

__global__ void knn_points_k(const float* __restrict__ proj, int* __restrict__ idx1) {
  int t = blockIdx.x * 256 + threadIdx.x;        // 4*2048
  float px = proj[t * 2], py = proj[t * 2 + 1];
  float bd[KNN]; int bi[KNN];
#pragma unroll
  for (int j = 0; j < KNN; ++j) { bd[j] = 3.0e38f; bi[j] = 0; }
  for (int g = 0; g < HW; ++g) {
    float dx = px - ((float)(g >> 5) + 0.5f);
    float dy = py - ((float)(g & 31) + 0.5f);
    float d = dx * dx + dy * dy;
    if (d < bd[KNN - 1]) {
      bd[KNN - 1] = d; bi[KNN - 1] = g;
#pragma unroll
      for (int j = KNN - 1; j > 0; --j) {
        if (bd[j] < bd[j - 1]) {
          float td = bd[j]; bd[j] = bd[j - 1]; bd[j - 1] = td;
          int   ti = bi[j]; bi[j] = bi[j - 1]; bi[j - 1] = ti;
        }
      }
    }
  }
#pragma unroll
  for (int j = 0; j < KNN; ++j) idx1[t * KNN + j] = bi[j];
}

__global__ void knn_pixels_k(const float* __restrict__ proj, int* __restrict__ idx2) {
  __shared__ float sp[L0 * 2];
  int obj = blockIdx.x >> 2;                     // 4 blocks per object
  for (int i = threadIdx.x; i < L0 * 2; i += 256) sp[i] = proj[obj * L0 * 2 + i];
  __syncthreads();
  int g = ((blockIdx.x & 3) << 8) + threadIdx.x;
  float gx = (float)(g >> 5) + 0.5f, gy = (float)(g & 31) + 0.5f;
  float bd[KNN]; int bi[KNN];
#pragma unroll
  for (int j = 0; j < KNN; ++j) { bd[j] = 3.0e38f; bi[j] = 0; }
  for (int p = 0; p < L0; ++p) {
    float dx = gx - sp[p * 2], dy = gy - sp[p * 2 + 1];
    float d = dx * dx + dy * dy;
    if (d < bd[KNN - 1]) {
      bd[KNN - 1] = d; bi[KNN - 1] = p;
#pragma unroll
      for (int j = KNN - 1; j > 0; --j) {
        if (bd[j] < bd[j - 1]) {
          float td = bd[j]; bd[j] = bd[j - 1]; bd[j - 1] = td;
          int   ti = bi[j]; bi[j] = bi[j - 1]; bi[j - 1] = ti;
        }
      }
    }
  }
#pragma unroll
  for (int j = 0; j < KNN; ++j) idx2[(obj * HW + g) * KNN + j] = bi[j];
}

// ---------------- WMMA GEMM ----------------
// out(rows x N) = epilogue( A(rows x K, optional row gather) * W(N x K)^T )
// MODE 0: +bias -> f32 | 1: relu((x+b)*BN*g+be) -> f16
// MODE 2: MODE1 then mean over 16 consecutive rows -> f16 | 3: sigmoid -> f32
#define TM 256
#define TN 64
#define TK 32
#define LSTR 40                   // 32 + 8 halfs pad: 80B rows, 16B aligned
#define ABUF (TM * LSTR)          // halfs per A buffer
#define BBUF (TN * LSTR)

template <int MODE, bool GATHER>
__global__ __launch_bounds__(256) void gemm_f16_k(
    const _Float16* __restrict__ A, const int* __restrict__ rowIdx,
    const _Float16* __restrict__ W, int K, int N,
    const float* __restrict__ bias, const float* __restrict__ gam,
    const float* __restrict__ bet, float* __restrict__ outF,
    _Float16* __restrict__ outH) {
  __shared__ _Float16 lA[2 * ABUF];
  __shared__ _Float16 lB[2 * BBUF];
  const int r0 = blockIdx.x * TM;
  const int n0 = blockIdx.y * TN;
  const int tid = threadIdx.x;
  const int lane = tid & 31;
  const int wave = tid >> 5;
  const int mBase = wave * 32;    // 2 M-subtiles per wave

  // ---- hoisted staging addresses (K-invariant) ----
  const int part = tid & 3;             // 8-half chunk within 32-half row
  const int rA = tid >> 2;              // 0..63
  const _Float16* aSrc[4];
  int aLds[4];
#pragma unroll
  for (int c = 0; c < 4; ++c) {
    int r = rA + c * 64;                // 0..255
    int srcRow = GATHER ? rowIdx[r0 + r] : (r0 + r);
    aSrc[c] = A + srcRow * K + part * 8;        // 32-bit offset math
    aLds[c] = r * LSTR + part * 8;
  }
  const _Float16* bSrc = W + (n0 + rA) * K + part * 8;
  const int bLds = rA * LSTR + part * 8;

  // ---- hoisted fragment LDS offsets ----
  const int fm = lane & 15, fkb = (lane >> 4) * 8;
  const int aOff = (mBase + fm) * LSTR + fkb;   // subtile 0; +16*LSTR for subtile 1
  const int fn = lane & 15, fkb2 = (lane >> 4) * 8;
  int bOff[4];
#pragma unroll
  for (int nt = 0; nt < 4; ++nt) bOff[nt] = (nt * 16 + fn) * LSTR + fkb2;

  v8f acc[2][4];
  const v8f zf = {0.f, 0.f, 0.f, 0.f, 0.f, 0.f, 0.f, 0.f};
#pragma unroll
  for (int s = 0; s < 2; ++s)
#pragma unroll
    for (int nt = 0; nt < 4; ++nt) acc[s][nt] = zf;

  const int nK = K / TK;

  auto stage = [&](int buf, int k0) {
#pragma unroll
    for (int c = 0; c < 4; ++c) {
#if USE_ASYNC_LDS
      __builtin_amdgcn_global_load_async_to_lds_b128(
          as1p(aSrc[c] + k0), as3p(&lA[buf * ABUF + aLds[c]]), 0, 0);
#else
      *(uint4*)&lA[buf * ABUF + aLds[c]] = *(const uint4*)(aSrc[c] + k0);
      __builtin_prefetch(aSrc[c] + k0 + TK, 0, 1);
#endif
    }
#if USE_ASYNC_LDS
    __builtin_amdgcn_global_load_async_to_lds_b128(
        as1p(bSrc + k0), as3p(&lB[buf * BBUF + bLds]), 0, 0);
#else
    *(uint4*)&lB[buf * BBUF + bLds] = *(const uint4*)(bSrc + k0);
    __builtin_prefetch(bSrc + k0 + TK, 0, 1);
#endif
  };

  auto fence_stage = [&]() {
#if USE_ASYNC_LDS
    wait_async0();
#endif
    __syncthreads();
  };

  stage(0, 0);
  fence_stage();

  for (int kt = 0; kt < nK; ++kt) {
    const int cur = kt & 1;
    if (kt + 1 < nK) stage(cur ^ 1, (kt + 1) * TK);

    // ---- preload ALL fragments for this K-step, then issue the 8 WMMAs ----
    const _Float16* bufA = lA + cur * ABUF;
    const _Float16* bufB = lB + cur * BBUF;
    v16h af[2];
    v16h bf[4];
#pragma unroll
    for (int s = 0; s < 2; ++s) {
      v8h alo = *(const v8h*)&bufA[aOff + s * 16 * LSTR];
      v8h ahi = *(const v8h*)&bufA[aOff + s * 16 * LSTR + 16];
      af[s] = __builtin_shufflevector(alo, ahi, 0, 1, 2, 3, 4, 5, 6, 7, 8, 9,
                                      10, 11, 12, 13, 14, 15);
    }
#pragma unroll
    for (int nt = 0; nt < 4; ++nt) {
      v8h blo = *(const v8h*)&bufB[bOff[nt]];
      v8h bhi = *(const v8h*)&bufB[bOff[nt] + 16];
      bf[nt] = __builtin_shufflevector(blo, bhi, 0, 1, 2, 3, 4, 5, 6, 7, 8, 9,
                                       10, 11, 12, 13, 14, 15);
    }
#pragma unroll
    for (int nt = 0; nt < 4; ++nt)
#pragma unroll
      for (int s = 0; s < 2; ++s)
        acc[s][nt] = __builtin_amdgcn_wmma_f32_16x16x32_f16(
            false, af[s], false, bf[nt], (short)0, acc[s][nt], false, false);

    if (kt + 1 < nK) fence_stage();
  }

  // epilogue — D layout: lane<16: N=lane, M=v; lane>=16: N=lane-16, M=8+v
  const int nl = lane & 15;
  const int mOff = (lane >> 4) * 8;
#pragma unroll
  for (int s = 0; s < 2; ++s) {
    const int rowBase = r0 + mBase + s * 16;
#pragma unroll
    for (int nt = 0; nt < 4; ++nt) {
      int nIdx = n0 + nt * 16 + nl;
      float b = bias[nIdx];
      if (MODE == 0) {
#pragma unroll
        for (int v = 0; v < 8; ++v)
          outF[(size_t)(rowBase + mOff + v) * N + nIdx] = acc[s][nt][v] + b;
      } else if (MODE == 1) {
        float gs = BN_SCALE * gam[nIdx], be = bet[nIdx];
#pragma unroll
        for (int v = 0; v < 8; ++v) {
          float val = fmaxf(0.f, (acc[s][nt][v] + b) * gs + be);
          outH[(size_t)(rowBase + mOff + v) * N + nIdx] = (_Float16)val;
        }
      } else if (MODE == 2) {
        float gs = BN_SCALE * gam[nIdx], be = bet[nIdx];
        float sum = 0.f;
#pragma unroll
        for (int v = 0; v < 8; ++v)
          sum += fmaxf(0.f, (acc[s][nt][v] + b) * gs + be);
        float tot = sum + __shfl_xor(sum, 16, 32);
        if (lane < 16)
          outH[(size_t)(rowBase >> 4) * N + nIdx] = (_Float16)(tot * (1.f / 16.f));
      } else {
#pragma unroll
        for (int v = 0; v < 8; ++v) {
          float x = acc[s][nt][v] + b;
          outF[(size_t)(rowBase + mOff + v) * N + nIdx] = 1.f / (1.f + __expf(-x));
        }
      }
    }
  }
}

// ---------------- GroupNorm stats / mask / merge ----------------

__global__ void col_stats_k(const float* __restrict__ X, int L, int N,
                            float* __restrict__ mu, float* __restrict__ rstd) {
  int col = blockIdx.x * 8 + (threadIdx.x >> 5);
  int lane = threadIdx.x & 31;
  if (col >= N) return;
  float s = 0.f, s2 = 0.f;
  for (int l = lane; l < L; l += 32) {
    float v = X[(size_t)l * N + col];
    s += v; s2 += v * v;
  }
#pragma unroll
  for (int off = 16; off; off >>= 1) {
    s += __shfl_xor(s, off, 32);
    s2 += __shfl_xor(s2, off, 32);
  }
  if (lane == 0) {
    float m = s / (float)L;
    float var = s2 / (float)L - m * m;
    mu[col] = m;
    rstd[col] = rsqrtf(var + EPSGN);
  }
}

__global__ void row_mask_k(const _Float16* __restrict__ Xh, int L, int N,
                           int* __restrict__ mask) {
  int row = blockIdx.x * 8 + (threadIdx.x >> 5);
  int lane = threadIdx.x & 31;
  if (row >= L) return;
  int any = 0;
  for (int n = lane; n < N; n += 32)
    any |= ((float)Xh[(size_t)row * N + n] != 0.f) ? 1 : 0;
#pragma unroll
  for (int off = 16; off; off >>= 1) any |= __shfl_xor(any, off, 32);
  if (lane == 0) mask[row] = any;
}

__global__ void merge_points_k(const float* __restrict__ Gp, const float* __restrict__ Gi,
                               const float* __restrict__ P, const float* __restrict__ I,
                               const float* __restrict__ muP, const float* __restrict__ rsP,
                               const float* __restrict__ muI, const float* __restrict__ rsI,
                               const float* __restrict__ ipg, const float* __restrict__ ipbe,
                               const float* __restrict__ iig, const float* __restrict__ iibe,
                               const int* __restrict__ mask, const float* __restrict__ points,
                               float* __restrict__ out, int obj) {
  int i = blockIdx.x * 256 + threadIdx.x;  // 2048*1024, l fastest
  int l = i & 2047, o = i >> 11;
  size_t idx = (size_t)l * 1024 + o;
  float gp = Gp[idx], gi = Gi[idx];
  float ipn = (P[idx] - muP[o]) * rsP[o] * ipg[o] + ipbe[o];
  float iin = (I[idx] - muI[o]) * rsI[o] * iig[o] + iibe[o];
  float fused = (gp * ipn + gi * iin) / (gp + gi);
  float raw = points[(size_t)o * 8192 + obj * 2048 + l];
  out[(size_t)o * 8192 + obj * 2048 + l] = mask[l] ? fused : raw;
}

__global__ void merge_pixels_k(const float* __restrict__ Gp, const float* __restrict__ Gi,
                               const float* __restrict__ P, const float* __restrict__ I,
                               const float* __restrict__ muP, const float* __restrict__ rsP,
                               const float* __restrict__ muI, const float* __restrict__ rsI,
                               const float* __restrict__ ipg, const float* __restrict__ ipbe,
                               const float* __restrict__ iig, const float* __restrict__ iibe,
                               const int* __restrict__ mask, const float* __restrict__ imgs,
                               float* __restrict__ out, int obj) {
  int i = blockIdx.x * 256 + threadIdx.x;  // 512*1024, l fastest
  int l = i & 1023, o = i >> 10;
  size_t idx = (size_t)l * 512 + o;
  float gp = Gp[idx], gi = Gi[idx];
  float ipn = (P[idx] - muP[o]) * rsP[o] * ipg[o] + ipbe[o];
  float iin = (I[idx] - muI[o]) * rsI[o] * iig[o] + iibe[o];
  float fused = (gp * ipn + gi * iin) / (gp + gi);
  float raw = imgs[(size_t)obj * 512 * 1024 + (size_t)o * 1024 + l];
  out[(size_t)obj * 512 * 1024 + (size_t)o * 1024 + l] = mask[l] ? fused : raw;
}

// ---------------- orchestration ----------------

extern "C" void kernel_launch(void* const* d_in, const int* in_sizes, int n_in,
                              void* d_out, int out_size, void* d_ws, size_t ws_size,
                              hipStream_t stream) {
  (void)in_sizes; (void)n_in; (void)out_size; (void)ws_size;
  const float* xys    = (const float*)d_in[0];
  const float* imgs   = (const float*)d_in[1];
  const float* points = (const float*)d_in[2];
  const float* Mm     = (const float*)d_in[3];
  const float* pce_w1 = (const float*)d_in[4];
  const float* pce_b1 = (const float*)d_in[5];
  const float* pce_g1 = (const float*)d_in[6];
  const float* pce_be1= (const float*)d_in[7];
  const float* pce_w2 = (const float*)d_in[8];
  const float* pce_b2 = (const float*)d_in[9];
  const float* pce_g2 = (const float*)d_in[10];
  const float* pce_be2= (const float*)d_in[11];
  const float* pxe_w1 = (const float*)d_in[12];
  const float* pxe_b1 = (const float*)d_in[13];
  const float* pxe_g1 = (const float*)d_in[14];
  const float* pxe_be1= (const float*)d_in[15];
  const float* pxe_w2 = (const float*)d_in[16];
  const float* pxe_b2 = (const float*)d_in[17];
  const float* pxe_g2 = (const float*)d_in[18];
  const float* pxe_be2= (const float*)d_in[19];
  const float* pcm_gpw = (const float*)d_in[20];
  const float* pcm_gpb = (const float*)d_in[21];
  const float* pcm_giw = (const float*)d_in[22];
  const float* pcm_gib = (const float*)d_in[23];
  const float* pcm_ipw = (const float*)d_in[24];
  const float* pcm_ipb = (const float*)d_in[25];
  const float* pcm_ipg = (const float*)d_in[26];
  const float* pcm_ipbe= (const float*)d_in[27];
  const float* pcm_iiw = (const float*)d_in[28];
  const float* pcm_iib = (const float*)d_in[29];
  const float* pcm_iig = (const float*)d_in[30];
  const float* pcm_iibe= (const float*)d_in[31];
  const float* pxm_gpw = (const float*)d_in[32];
  const float* pxm_gpb = (const float*)d_in[33];
  const float* pxm_giw = (const float*)d_in[34];
  const float* pxm_gib = (const float*)d_in[35];
  const float* pxm_ipw = (const float*)d_in[36];
  const float* pxm_ipb = (const float*)d_in[37];
  const float* pxm_ipg = (const float*)d_in[38];
  const float* pxm_ipbe= (const float*)d_in[39];
  const float* pxm_iiw = (const float*)d_in[40];
  const float* pxm_iib = (const float*)d_in[41];
  const float* pxm_iig = (const float*)d_in[42];
  const float* pxm_iibe= (const float*)d_in[43];

  float* fiOut = (float*)d_out;                       // (4,512,1024)
  float* fpOut = fiOut + (size_t)NOBJ * CIMG * HW;    // (1024,8192)

  char* wp = (char*)d_ws;
  auto alloc = [&](size_t bytes) -> void* {
    void* p = (void*)wp;
    wp += (bytes + 255) & ~(size_t)255;
    return p;
  };

  _Float16* imgT = (_Float16*)alloc((size_t)NOBJ * HW * CIMG * 2);
  _Float16* ptsT = (_Float16*)alloc((size_t)NOBJ * L0 * DPTS * 2);
  _Float16* w_pce1 = (_Float16*)alloc(512 * 512 * 2);
  _Float16* w_pce2 = (_Float16*)alloc(1024 * 512 * 2);
  _Float16* w_pxe1 = (_Float16*)alloc(1024 * 1024 * 2);
  _Float16* w_pxe2 = (_Float16*)alloc(512 * 1024 * 2);
  _Float16* w_pcm_gp = (_Float16*)alloc(1024 * 1024 * 2);
  _Float16* w_pcm_gi = (_Float16*)alloc(1024 * 1024 * 2);
  _Float16* w_pcm_ip = (_Float16*)alloc(1024 * 1024 * 2);
  _Float16* w_pcm_ii = (_Float16*)alloc(1024 * 1024 * 2);
  _Float16* w_pxm_gp = (_Float16*)alloc(512 * 512 * 2);
  _Float16* w_pxm_gi = (_Float16*)alloc(512 * 512 * 2);
  _Float16* w_pxm_ip = (_Float16*)alloc(512 * 512 * 2);
  _Float16* w_pxm_ii = (_Float16*)alloc(512 * 512 * 2);
  float* proj = (float*)alloc((size_t)NOBJ * L0 * 2 * 4);
  int* idx1 = (int*)alloc((size_t)NOBJ * L0 * KNN * 4);
  int* idx2 = (int*)alloc((size_t)NOBJ * HW * KNN * 4);
  _Float16* h1 = (_Float16*)alloc((size_t)32768 * 512 * 2);  // == 16384*1024
  _Float16* appear = (_Float16*)alloc((size_t)L0 * 1024 * 2);
  _Float16* extb = (_Float16*)alloc((size_t)HW * 512 * 2);
  float* Gp = (float*)alloc((size_t)L0 * 1024 * 4);
  float* Gi = (float*)alloc((size_t)L0 * 1024 * 4);
  float* Pb = (float*)alloc((size_t)L0 * 1024 * 4);
  float* Ib = (float*)alloc((size_t)L0 * 1024 * 4);
  float* muP = (float*)alloc(1024 * 4);
  float* rsP = (float*)alloc(1024 * 4);
  float* muI = (float*)alloc(1024 * 4);
  float* rsI = (float*)alloc(1024 * 4);
  int* mask = (int*)alloc(L0 * 4);

  auto cast = [&](const float* s, _Float16* d, int n) {
    cast_f16_k<<<(n + 255) / 256, 256, 0, stream>>>(s, d, n);
  };
  cast(pce_w1, w_pce1, 512 * 512);
  cast(pce_w2, w_pce2, 1024 * 512);
  cast(pxe_w1, w_pxe1, 1024 * 1024);
  cast(pxe_w2, w_pxe2, 512 * 1024);
  cast(pcm_gpw, w_pcm_gp, 1024 * 1024);
  cast(pcm_giw, w_pcm_gi, 1024 * 1024);
  cast(pcm_ipw, w_pcm_ip, 1024 * 1024);
  cast(pcm_iiw, w_pcm_ii, 1024 * 1024);
  cast(pxm_gpw, w_pxm_gp, 512 * 512);
  cast(pxm_giw, w_pxm_gi, 512 * 512);
  cast(pxm_ipw, w_pxm_ip, 512 * 512);
  cast(pxm_iiw, w_pxm_ii, 512 * 512);

  transpose_img_k<<<(NOBJ * HW * CIMG) / 256, 256, 0, stream>>>(imgs, imgT);
  transpose_pts_k<<<(NOBJ * L0 * DPTS) / 256, 256, 0, stream>>>(points, ptsT);
  project_k<<<(NOBJ * L0) / 256, 256, 0, stream>>>(xys, Mm, proj);
  knn_points_k<<<(NOBJ * L0) / 256, 256, 0, stream>>>(proj, idx1);
  knn_pixels_k<<<(NOBJ * HW) / 256, 256, 0, stream>>>(proj, idx2);

  auto gemm = [&](int mode, const _Float16* A, const int* ridx, const _Float16* Wm,
                  int rows, int K, int N, const float* bias, const float* ga,
                  const float* be, float* oF, _Float16* oH) {
    dim3 g(rows / TM, N / TN);
    if (ridx) {
      switch (mode) {
        case 0: gemm_f16_k<0, true><<<g, 256, 0, stream>>>(A, ridx, Wm, K, N, bias, ga, be, oF, oH); break;
        case 1: gemm_f16_k<1, true><<<g, 256, 0, stream>>>(A, ridx, Wm, K, N, bias, ga, be, oF, oH); break;
        case 2: gemm_f16_k<2, true><<<g, 256, 0, stream>>>(A, ridx, Wm, K, N, bias, ga, be, oF, oH); break;
        default: gemm_f16_k<3, true><<<g, 256, 0, stream>>>(A, ridx, Wm, K, N, bias, ga, be, oF, oH); break;
      }
    } else {
      switch (mode) {
        case 0: gemm_f16_k<0, false><<<g, 256, 0, stream>>>(A, ridx, Wm, K, N, bias, ga, be, oF, oH); break;
        case 1: gemm_f16_k<1, false><<<g, 256, 0, stream>>>(A, ridx, Wm, K, N, bias, ga, be, oF, oH); break;
        case 2: gemm_f16_k<2, false><<<g, 256, 0, stream>>>(A, ridx, Wm, K, N, bias, ga, be, oF, oH); break;
        default: gemm_f16_k<3, false><<<g, 256, 0, stream>>>(A, ridx, Wm, K, N, bias, ga, be, oF, oH); break;
      }
    }
  };

  for (int obj = 0; obj < NOBJ; ++obj) {
    const _Float16* imgT_o = imgT + (size_t)obj * HW * CIMG;
    const _Float16* ptsT_o = ptsT + (size_t)obj * L0 * DPTS;
    const int* idx1_o = idx1 + (size_t)obj * L0 * KNN;
    const int* idx2_o = idx2 + (size_t)obj * HW * KNN;

    // points-centered appearance: gather(imgT) -> 512 -> 512 -> mean16 -> 1024
    gemm(1, imgT_o, idx1_o, w_pce1, L0 * KNN, 512, 512, pce_b1, pce_g1, pce_be1, nullptr, h1);
    gemm(2, h1, nullptr, w_pce2, L0 * KNN, 512, 1024, pce_b2, pce_g2, pce_be2, nullptr, appear);

    // pixel-centered geometry: gather(ptsT) -> 1024 -> 1024 -> mean16 -> 512
    gemm(1, ptsT_o, idx2_o, w_pxe1, HW * KNN, 1024, 1024, pxe_b1, pxe_g1, pxe_be1, nullptr, h1);
    gemm(2, h1, nullptr, w_pxe2, HW * KNN, 1024, 512, pxe_b2, pxe_g2, pxe_be2, nullptr, extb);

    // point fusion (C=1024, L=2048)
    gemm(3, ptsT_o, nullptr, w_pcm_gp, L0, 1024, 1024, pcm_gpb, nullptr, nullptr, Gp, nullptr);
    gemm(3, appear, nullptr, w_pcm_gi, L0, 1024, 1024, pcm_gib, nullptr, nullptr, Gi, nullptr);
    gemm(0, ptsT_o, nullptr, w_pcm_ip, L0, 1024, 1024, pcm_ipb, nullptr, nullptr, Pb, nullptr);
    gemm(0, appear, nullptr, w_pcm_ii, L0, 1024, 1024, pcm_iib, nullptr, nullptr, Ib, nullptr);
    col_stats_k<<<1024 / 8, 256, 0, stream>>>(Pb, L0, 1024, muP, rsP);
    col_stats_k<<<1024 / 8, 256, 0, stream>>>(Ib, L0, 1024, muI, rsI);
    row_mask_k<<<L0 / 8, 256, 0, stream>>>(appear, L0, 1024, mask);
    merge_points_k<<<(L0 * 1024) / 256, 256, 0, stream>>>(
        Gp, Gi, Pb, Ib, muP, rsP, muI, rsI, pcm_ipg, pcm_ipbe, pcm_iig, pcm_iibe,
        mask, points, fpOut, obj);

    // pixel fusion (C=512, L=1024)
    gemm(3, imgT_o, nullptr, w_pxm_gp, HW, 512, 512, pxm_gpb, nullptr, nullptr, Gp, nullptr);
    gemm(3, extb, nullptr, w_pxm_gi, HW, 512, 512, pxm_gib, nullptr, nullptr, Gi, nullptr);
    gemm(0, imgT_o, nullptr, w_pxm_ip, HW, 512, 512, pxm_ipb, nullptr, nullptr, Pb, nullptr);
    gemm(0, extb, nullptr, w_pxm_ii, HW, 512, 512, pxm_iib, nullptr, nullptr, Ib, nullptr);
    col_stats_k<<<512 / 8, 256, 0, stream>>>(Pb, HW, 512, muP, rsP);
    col_stats_k<<<512 / 8, 256, 0, stream>>>(Ib, HW, 512, muI, rsI);
    row_mask_k<<<HW / 8, 256, 0, stream>>>(extb, HW, 512, mask);
    merge_pixels_k<<<(HW * 512) / 256, 256, 0, stream>>>(
        Gp, Gi, Pb, Ib, muP, rsP, muI, rsI, pxm_ipg, pxm_ipbe, pxm_iig, pxm_iibe,
        mask, imgs, fiOut, obj);
  }
}
